// TritonLinearETRow_34626026340515
// MI455X (gfx1250) — matrix operand order repaired and checked by reading, not compiled, not run on hardware
//
#include <hip/hip_runtime.h>
#include <stdint.h>

typedef __attribute__((ext_vector_type(16))) _Float16 v16h;
typedef __attribute__((ext_vector_type(8)))  _Float16 v8h;
typedef __attribute__((ext_vector_type(4)))  _Float16 v4h;
typedef __attribute__((ext_vector_type(8)))  float    v8f;
typedef __attribute__((ext_vector_type(4)))  float    v4f;

#define KC      32          // K-chunk per stage (one wmma k-step)
#define LDA32   36          // f32 staging row stride (KC+4, keeps 16B chunk alignment)
#define LDA16   40          // f16 row stride in halfs (mult. of 8 -> 16B aligned chunks)
#define M_TILE  128
#define N_TILE  128
#define KDIM    1024
#define NDIM    1024

// CDNA5 async global -> LDS copy (16B), tracked by ASYNCcnt.
__device__ __forceinline__ void async_copy_b128(uint32_t lds_off, const float* gptr) {
  asm volatile("global_load_async_to_lds_b128 %0, %1, off"
               :: "v"(lds_off), "v"(gptr)
               : "memory");
}

__global__ __launch_bounds__(256)
void linear_et_row_kernel(const float* __restrict__ x,
                          const float* __restrict__ w,
                          const float* __restrict__ bias,
                          float* __restrict__ out) {
  // f32 async staging (double buffered) + packed f16 operand tiles (double buffered)
  __shared__ float    As32[2][M_TILE][LDA32];
  __shared__ float    Bs32[2][N_TILE][LDA32];
  __shared__ _Float16 As16[2][M_TILE][LDA16];
  __shared__ _Float16 Bs16[2][N_TILE][LDA16];

  const int tid  = threadIdx.x;
  const int lane = tid & 31;
  const int wave = tid >> 5;
  const int wm   = wave & 3;    // 4 x 32-row slabs   (wave tile: 32 rows)
  const int wn   = wave >> 2;   // 2 x 64-col slabs   (wave tile: 64 cols)
  const int g16  = lane >> 4;   // lane half-group (0/1)
  const int l16  = lane & 15;

  const int m0 = blockIdx.y * M_TILE;
  const int n0 = blockIdx.x * N_TILE;

  uint32_t asBase[2], bsBase[2];
  asBase[0] = (uint32_t)(uintptr_t)&As32[0][0][0];
  asBase[1] = (uint32_t)(uintptr_t)&As32[1][0][0];
  bsBase[0] = (uint32_t)(uintptr_t)&Bs32[0][0][0];
  bsBase[1] = (uint32_t)(uintptr_t)&Bs32[1][0][0];

  // Issue one stage: A 128x32 f32 = 1024 b128 chunks (4/thread),
  //                  B 128x32 f32 = 1024 b128 chunks (4/thread).
  auto issue_stage = [&](int kt, int buf) {
    const int k0 = kt * KC;
#pragma unroll
    for (int i = 0; i < 4; ++i) {
      const int q = tid + 256 * i, row = q >> 3, ch = q & 7;
      async_copy_b128(asBase[buf] + (uint32_t)((row * LDA32 + ch * 4) * 4),
                      x + (size_t)(m0 + row) * KDIM + k0 + ch * 4);
    }
#pragma unroll
    for (int i = 0; i < 4; ++i) {
      const int q = tid + 256 * i, row = q >> 3, ch = q & 7;
      async_copy_b128(bsBase[buf] + (uint32_t)((row * LDA32 + ch * 4) * 4),
                      w + (size_t)(n0 + row) * KDIM + k0 + ch * 4);
    }
  };

  // Cooperative f32 -> f16 conversion of one staged tile (each value converted once).
  auto convert_stage = [&](int buf) {
#pragma unroll
    for (int i = 0; i < 4; ++i) {
      const int q = tid + 256 * i, row = q >> 3, ch = q & 7;
      v4f f = *(const v4f*)&As32[buf][row][ch * 4];
      v4h h;
      h.x = (_Float16)f.x; h.y = (_Float16)f.y;
      h.z = (_Float16)f.z; h.w = (_Float16)f.w;
      *(v4h*)&As16[buf][row][ch * 4] = h;
    }
#pragma unroll
    for (int i = 0; i < 4; ++i) {
      const int q = tid + 256 * i, row = q >> 3, ch = q & 7;
      v4f f = *(const v4f*)&Bs32[buf][row][ch * 4];
      v4h h;
      h.x = (_Float16)f.x; h.y = (_Float16)f.y;
      h.z = (_Float16)f.z; h.w = (_Float16)f.w;
      *(v4h*)&Bs16[buf][row][ch * 4] = h;
    }
  };

  v8f acc[2][4] = {};

  issue_stage(0, 0);   // 2-deep async prefetch pipeline
  issue_stage(1, 1);

  const int NKT = KDIM / KC;  // 32 stages
  for (int kt = 0; kt < NKT; ++kt) {
    const int buf = kt & 1;
    if (kt + 1 < NKT) asm volatile("s_wait_asynccnt 0x8" ::: "memory");
    else              asm volatile("s_wait_asynccnt 0x0" ::: "memory");
    __syncthreads();             // stage kt f32 tile visible to all waves

    convert_stage(buf);
    __syncthreads();             // f16 tile ready; f32 buffer free for reuse

    if (kt + 2 < NKT) issue_stage(kt + 2, buf);

    // A fragments: 16-bit A 16x32 layout
    //   lane grp 0: halfs 0..7 = K0..7,  8..15 = K16..23
    //   lane grp 1: halfs 0..7 = K8..15, 8..15 = K24..31
    v16h a[2];
#pragma unroll
    for (int i = 0; i < 2; ++i) {
      const _Float16* ap = &As16[buf][wm * 32 + i * 16 + l16][0];
      v8h lo = *(const v8h*)(ap + g16 * 8);
      v8h hi = *(const v8h*)(ap + 16 + g16 * 8);
      a[i] = __builtin_shufflevector(lo, hi, 0, 1, 2, 3, 4, 5, 6, 7,
                                             8, 9, 10, 11, 12, 13, 14, 15);
    }
    // B fragments: 32x16 layout, lane = column n (= weight row),
    //   grp 0: halfs 0..7 = K0..7, 8..15 = K16..23; grp 1: K8..15, K24..31
    v16h b[4];
#pragma unroll
    for (int j = 0; j < 4; ++j) {
      const _Float16* bp = &Bs16[buf][wn * 64 + j * 16 + l16][g16 * 8];
      v8h lo = *(const v8h*)(bp);
      v8h hi = *(const v8h*)(bp + 16);
      b[j] = __builtin_shufflevector(lo, hi, 0, 1, 2, 3, 4, 5, 6, 7,
                                             8, 9, 10, 11, 12, 13, 14, 15);
    }

#pragma unroll
    for (int i = 0; i < 2; ++i)
#pragma unroll
      for (int j = 0; j < 4; ++j)
        acc[i][j] = __builtin_amdgcn_wmma_f32_16x16x32_f16(
            false, a[i], false, b[j], (short)0, acc[i][j], false, false);
  }

  // Epilogue: C 16x16 f32 layout: VGPR r, lanes 0-15 -> M=r, lanes 16-31 -> M=r+8, N=l16
#pragma unroll
  for (int j = 0; j < 4; ++j) {
    const int gn = n0 + wn * 64 + j * 16 + l16;
    const float bv = bias[gn];
#pragma unroll
    for (int i = 0; i < 2; ++i) {
      const int mb = m0 + wm * 32 + i * 16 + g16 * 8;
#pragma unroll
      for (int r = 0; r < 8; ++r)
        out[(size_t)(mb + r) * NDIM + gn] = acc[i][j][r] + bv;
    }
  }
}

extern "C" void kernel_launch(void* const* d_in, const int* in_sizes, int n_in,
                              void* d_out, int out_size, void* d_ws, size_t ws_size,
                              hipStream_t stream) {
  const float* x  = (const float*)d_in[0];   // [M, 1024] f32
  const float* w  = (const float*)d_in[1];   // [1024, 1024] f32
  const float* bs = (const float*)d_in[2];   // [1024] f32
  float* out = (float*)d_out;                // [M, 1024] f32

  const int M = in_sizes[0] / KDIM;          // 32768
  dim3 grid(NDIM / N_TILE, M / M_TILE);      // (8, 256)
  linear_et_row_kernel<<<grid, 256, 0, stream>>>(x, w, bs, out);
}
